// SingleViewNet_78537771975378
// MI455X (gfx1250) — matrix operand order repaired and checked by reading, not compiled, not run on hardware
//
#include <hip/hip_runtime.h>
#include <hip/hip_bf16.h>

// ---------------------------------------------------------------------------
// SingleViewNet (PointPillars-style) for MI455X / gfx1250, wave32 + WMMA f16.
// All convs/deconvs = implicit GEMM with V_WMMA_F32_16X16X32_F16.
// Activations NHWC f16 (lane A-fragment == two aligned b128 loads),
// weights repacked on-device into per-lane B-fragment order and staged
// per-tap into LDS (shared by the 4 waves of a block).
// ---------------------------------------------------------------------------

typedef __attribute__((ext_vector_type(16))) _Float16 v16h;
typedef __attribute__((ext_vector_type(8)))  float    v8f;

#define PC0 0.0f
#define PC1 (-39.68f)
#define PC3 69.12f
#define PC4 39.68f
#define VS0 0.32f
#define VS1 0.32f
#define NXc 216
#define NYc 248
#define Cc  64
#define INC 16
#define Bc  4
#define Nc  300000
#define EPSc 1e-3f

union AFrag { v16h v; uint4 u[2]; };

// ---------------------------------------------------------------------------
// 0) utility kernels
// ---------------------------------------------------------------------------
__global__ void zero_f32_kernel(float* __restrict__ p, long n) {
    long i = (long)blockIdx.x * blockDim.x + threadIdx.x;
    long stride = (long)gridDim.x * blockDim.x;
    for (; i < n; i += stride) p[i] = 0.0f;
}

__global__ void f32_to_f16_kernel(const float* __restrict__ s, _Float16* __restrict__ d, long n) {
    long i = (long)blockIdx.x * blockDim.x + threadIdx.x;
    long stride = (long)gridDim.x * blockDim.x;
    for (; i < n; i += stride) d[i] = (_Float16)s[i];
}

// ---------------------------------------------------------------------------
// 1) PointNet (relu(bn1d(feat @ W))) fused with scatter-max into NHWC canvas.
//    Post-ReLU values are >= 0, canvas zero-initialized -> uint atomicMax is
//    exactly segment_max with the -inf->0 replacement of the reference.
// ---------------------------------------------------------------------------
__global__ __launch_bounds__(256) void pointnet_scatter_kernel(
    const float* __restrict__ xyz, const float* __restrict__ feat,
    const int* __restrict__ bidx, const float* __restrict__ W,
    const float* __restrict__ gamma, const float* __restrict__ beta,
    const float* __restrict__ mean, const float* __restrict__ var,
    float* __restrict__ canvas /* [B,NY,NX,64] f32, zeroed */)
{
    __shared__ float sW[INC * Cc];
    __shared__ float sScale[Cc], sShift[Cc];
    int tid = threadIdx.x;
    for (int i = tid; i < INC * Cc; i += 256) sW[i] = W[i];
    if (tid < Cc) {
        float s = gamma[tid] * rsqrtf(var[tid] + EPSc);
        sScale[tid] = s;
        sShift[tid] = beta[tid] - mean[tid] * s;
    }
    __syncthreads();

    int p = blockIdx.x * 256 + tid;
    if (p >= Nc) return;

    float f[INC];
#pragma unroll
    for (int i = 0; i < INC; ++i) f[i] = feat[(long)p * INC + i];

    float x = xyz[(long)p * 3 + 0];
    float y = xyz[(long)p * 3 + 1];
    int b = bidx[p];
    int xc = (int)floorf((x - PC0) / VS0); xc = min(max(xc, 0), NXc - 1);
    int yc = (int)floorf((y - PC1) / VS1); yc = min(max(yc, 0), NYc - 1);

    unsigned int* cell = (unsigned int*)canvas + (((long)b * NYc + yc) * NXc + xc) * Cc;
#pragma unroll 4
    for (int o = 0; o < Cc; ++o) {
        float acc = 0.f;
#pragma unroll
        for (int i = 0; i < INC; ++i) acc += f[i] * sW[i * Cc + o];
        float v = fmaxf(acc * sScale[o] + sShift[o], 0.f);
        atomicMax(cell + o, __float_as_uint(v));  // v >= 0: uint order == float order
    }
}

// ---------------------------------------------------------------------------
// 2) Weight repack: OIHW/IOHW fp32 -> per-lane WMMA B-fragment order (f16).
//    Layout: [tap][chunk][ntile][lane(32)][16 halves], where per ISA 7.12.2
//    B(32x16,f16): lane = column n (l&15), half = l>>4, element e -> K=8*half+e (e<8), 16+8*half+(e-8) (e>=8).
// ---------------------------------------------------------------------------
__global__ void repack_weights_kernel(
    const float* __restrict__ src, _Float16* __restrict__ dst,
    int Cout, int Cin, int KH, int KW, int iohw)
{
    int nchunk = Cin >> 5;
    int total = KH * KW * nchunk * 4 * 512;
    int idx = blockIdx.x * blockDim.x + threadIdx.x;
    if (idx >= total) return;
    int e = idx & 15;
    int r = idx >> 4;
    int lane = r & 31; r >>= 5;
    int t = r & 3;     r >>= 2;
    int c = r % nchunk;
    int tap = r / nchunk;
    int kh = tap / KW, kw = tap - kh * KW;
    int n = t * 16 + (lane & 15);
    int k = (e < 8) ? (8 * (lane >> 4) + e) : (16 + 8 * (lane >> 4) + (e - 8));
    int cin = c * 32 + k;
    float v = iohw ? src[(((long)cin * Cout + n) * KH + kh) * KW + kw]
                   : src[(((long)n * Cin + cin) * KH + kh) * KW + kw];
    dst[idx] = (_Float16)v;
}

// ---------------------------------------------------------------------------
// 3) Implicit-GEMM conv via WMMA. 128 threads = 4 waves; each wave owns a
//    16-pixel x 64-channel output tile (4 v8f accumulators). NHWC f16 input:
//    A-fragment per lane (ISA 16-bit A 16x32 layout) = two aligned b128 loads
//    at channel offsets 8*half and 16+8*half within the 32-wide K chunk.
//    B fragments are staged per-tap into LDS (one copy per block, 4 waves
//    share) and read back as ds_load_b128. NCHUNK is a template constant so
//    the K loop fully unrolls into back-to-back WMMAs.
//    Epilogue fuses BN (or bias), residual add, ReLU; writes f16 NHWC slice
//    (stride/offset mapping for deconv sub-tap scatter) and/or dense f32 NHWC.
// ---------------------------------------------------------------------------
template <int NCHUNK>
__global__ __launch_bounds__(128) void conv_wmma_kernel(
    const _Float16* __restrict__ in, int Hin, int Win,
    const _Float16* __restrict__ wpk, int KH, int KW, int stride, int pad,
    int Hout, int Wout,
    const float* __restrict__ gamma, const float* __restrict__ beta,
    const float* __restrict__ mean,  const float* __restrict__ var,
    const float* __restrict__ bias,
    const _Float16* __restrict__ resid, int resC,
    int relu,
    _Float16* __restrict__ outh, int outC, int outc0, int OH, int OW,
    int osy, int osx, int oy0, int ox0,
    float* __restrict__ outf)
{
    constexpr int Cin = NCHUNK * 32;
    __shared__ _Float16 sB[NCHUNK * 2048];  // one tap: [chunk][ntile][lane][16]

    const int lane = threadIdx.x & 31;
    const int wave = threadIdx.x >> 5;
    const int b    = blockIdx.y;
    const int npix = Hout * Wout;
    const int pixbase = blockIdx.x * 64 + wave * 16;
    const int mrow = lane & 15;
    const int half = lane >> 4;

    const _Float16* inb = in + (long)b * Hin * Win * Cin;

    v8f acc0 = {}, acc1 = {}, acc2 = {}, acc3 = {};

    // Per-lane A-row coordinates (row M = mrow of the 16-pixel tile).
    int pa = pixbase + mrow;
    bool pav = pa < npix;
    int aoy = pav ? (pa / Wout) : 0;
    int aox = pav ? (pa - aoy * Wout) : 0;

    const v16h* bsh = (const v16h*)sB;
    const int ntap = KH * KW;

    for (int tap = 0; tap < ntap; ++tap) {
        // Stage this tap's B fragments into LDS (all 128 threads cooperate).
        __syncthreads();  // previous tap's reads complete
        {
            const uint4* wsrc = (const uint4*)(wpk + (long)tap * NCHUNK * 2048);
            uint4* wdst = (uint4*)sB;
#pragma unroll
            for (int i = 0; i < NCHUNK * 2; ++i)
                wdst[i * 128 + threadIdx.x] = wsrc[i * 128 + threadIdx.x];
        }
        __syncthreads();

        int kh = tap / KW, kw = tap - kh * KW;
        int iy = aoy * stride - pad + kh;
        int ix = aox * stride - pad + kw;
        bool ok = pav && iy >= 0 && iy < Hin && ix >= 0 && ix < Win;
        const _Float16* abase = inb + ((long)iy * Win + ix) * Cin + half * 8;
        if (ok) __builtin_prefetch(abase + (long)Win * Cin, 0, 1);  // global_prefetch_b8

#pragma unroll
        for (int c = 0; c < NCHUNK; ++c) {
            AFrag a;
            if (ok) {
                a.u[0] = *(const uint4*)(abase + c * 32);        // K = 8h + 0..7
                a.u[1] = *(const uint4*)(abase + c * 32 + 16);   // K = 16 + 8h + 0..7
            } else {
                a.u[0] = make_uint4(0u, 0u, 0u, 0u);
                a.u[1] = make_uint4(0u, 0u, 0u, 0u);
            }
            v16h b0 = bsh[(c * 4 + 0) * 32 + lane];
            v16h b1 = bsh[(c * 4 + 1) * 32 + lane];
            v16h b2 = bsh[(c * 4 + 2) * 32 + lane];
            v16h b3 = bsh[(c * 4 + 3) * 32 + lane];
            // EXEC is all-ones here (no early returns in this kernel).
            acc0 = __builtin_amdgcn_wmma_f32_16x16x32_f16(false, a.v, false, b0, (short)0, acc0, false, false);
            acc1 = __builtin_amdgcn_wmma_f32_16x16x32_f16(false, a.v, false, b1, (short)0, acc1, false, false);
            acc2 = __builtin_amdgcn_wmma_f32_16x16x32_f16(false, a.v, false, b2, (short)0, acc2, false, false);
            acc3 = __builtin_amdgcn_wmma_f32_16x16x32_f16(false, a.v, false, b3, (short)0, acc3, false, false);
        }
    }

    // Epilogue: D layout (32-bit C/D 16x16): element v -> M = v + 8*half, N = mrow.
    auto epi = [&](v8f acc, int t) {
        int n = t * 16 + mrow;
        float sc = 1.f, sh = 0.f;
        if (gamma) {
            float s = gamma[n] * rsqrtf(var[n] + EPSc);
            sc = s; sh = beta[n] - mean[n] * s;
        } else if (bias) {
            sh = bias[n];
        }
#pragma unroll
        for (int v = 0; v < 8; ++v) {
            int p = pixbase + half * 8 + v;
            if (p >= npix) continue;
            int py = p / Wout, px = p - py * Wout;
            float val = acc[v] * sc + sh;
            if (resid) val += (float)resid[(((long)b * Hout + py) * Wout + px) * resC + n];
            if (relu) val = fmaxf(val, 0.f);
            if (outh) outh[(((long)b * OH + (oy0 + py * osy)) * OW + (ox0 + px * osx)) * outC + outc0 + n] = (_Float16)val;
            if (outf) outf[(((long)b * Hout + py) * Wout + px) * 64 + n] = val;
        }
    };
    epi(acc0, 0); epi(acc1, 1); epi(acc2, 2); epi(acc3, 3);
}

// ---------------------------------------------------------------------------
// 4) Bilinear grid-sample of the final f32 NHWC BEV map at the point coords.
// ---------------------------------------------------------------------------
__global__ __launch_bounds__(256) void grid_sample_kernel(
    const float* __restrict__ xyz, const int* __restrict__ bidx,
    const float* __restrict__ bev /* [B,248,216,64] */, float* __restrict__ out)
{
    long idx = (long)blockIdx.x * 256 + threadIdx.x;
    if (idx >= (long)Nc * Cc) return;
    int p = (int)(idx >> 6);
    int c = (int)(idx & 63);
    float x = xyz[(long)p * 3 + 0];
    float y = xyz[(long)p * 3 + 1];
    float gx = (x - PC0) / ((PC3 - PC0) * 0.5f) - 1.0f;
    float gy = (y - PC1) / ((PC4 - PC1) * 0.5f) - 1.0f;
    float xp = (gx + 1.0f) * (float)NXc * 0.5f - 0.5f;
    float yp = (gy + 1.0f) * (float)NYc * 0.5f - 0.5f;
    float x0f = floorf(xp), y0f = floorf(yp);
    float wx = xp - x0f, wy = yp - y0f;
    int x0 = (int)x0f, y0 = (int)y0f;
    int b = bidx[p];
    const float* base = bev + (long)b * NYc * NXc * Cc + c;
    auto g = [&](int yi, int xi) -> float {
        if (yi < 0 || yi >= NYc || xi < 0 || xi >= NXc) return 0.f;
        return base[((long)yi * NXc + xi) * Cc];
    };
    out[idx] = g(y0, x0)       * (1.f - wx) * (1.f - wy)
             + g(y0, x0 + 1)   * wx         * (1.f - wy)
             + g(y0 + 1, x0)   * (1.f - wx) * wy
             + g(y0 + 1, x0 + 1) * wx * wy;
}

// ---------------------------------------------------------------------------
// Host side
// ---------------------------------------------------------------------------

// Input indices assuming pytree flattening with sorted dict keys at each level
// after the three positional arrays (pts_xyz, pts_features, batch_idx).
// params: conv_b, conv_w, deconv2_w, deconv3_w, pointnet_bn{beta,gamma,mean,var},
//         pointnet_w, res1{bn1,bn2,conv1,conv2}, res2{bn1,bn2,conv1,conv2,ds_bn,ds_conv}, res3{...}
enum {
    IN_XYZ = 0, IN_FEAT = 1, IN_BIDX = 2,
    IN_CONV_B = 3, IN_CONV_W = 4, IN_DC2_W = 5, IN_DC3_W = 6,
    IN_PN_BETA = 7, /* gamma=8 mean=9 var=10 */
    IN_PN_W = 11,
    IN_R1_BN1 = 12, IN_R1_BN2 = 16, IN_R1_C1 = 20, IN_R1_C2 = 21,
    IN_R2_BN1 = 22, IN_R2_BN2 = 26, IN_R2_C1 = 30, IN_R2_C2 = 31,
    IN_R2_DSBN = 32, IN_R2_DS = 36,
    IN_R3_BN1 = 37, IN_R3_BN2 = 41, IN_R3_C1 = 45, IN_R3_C2 = 46,
    IN_R3_DSBN = 47, IN_R3_DS = 51,
};

// Workspace layout (bytes)
static constexpr long SZ_CANVAS_F32 = (long)Bc * NYc * NXc * Cc * 4;      // 54,853,632
static constexpr long SZ_CANVAS_F16 = (long)Bc * NYc * NXc * Cc * 2;
static constexpr long SZ_TBUF       = SZ_CANVAS_F16;                      // reused t1/t2/t3
static constexpr long SZ_L2BUF      = (long)Bc * 124 * 108 * Cc * 2;      // 6,856,704
static constexpr long SZ_L3BUF      = (long)Bc * 62 * 54 * Cc * 2;        // 1,714,176
static constexpr long SZ_CONCAT     = (long)Bc * NYc * NXc * 192 * 2;     // 82,280,448
static constexpr long SZ_FINAL      = (long)Bc * NYc * NXc * Cc * 4;

static constexpr long OFF_CANVAS_F32 = 0;
static constexpr long OFF_CANVAS_F16 = OFF_CANVAS_F32 + SZ_CANVAS_F32;
static constexpr long OFF_TBUF       = OFF_CANVAS_F16 + SZ_CANVAS_F16;
static constexpr long OFF_DSBUF      = OFF_TBUF + SZ_TBUF;
static constexpr long OFF_OUT2       = OFF_DSBUF + SZ_L2BUF;
static constexpr long OFF_OUT3       = OFF_OUT2 + SZ_L2BUF;
static constexpr long OFF_CONCAT     = OFF_OUT3 + SZ_L3BUF;
static constexpr long OFF_FINAL      = OFF_CONCAT + SZ_CONCAT;
static constexpr long OFF_WPK        = OFF_FINAL + SZ_FINAL;

// Packed-weight sub-offsets (in _Float16 elements)
static constexpr long W_R1C1 = 0;       // 36864 each for 3x3 64->64
static constexpr long W_R1C2 = 36864;
static constexpr long W_R2C1 = 73728;
static constexpr long W_R2C2 = 110592;
static constexpr long W_R2DS = 147456;  // 4096 (1x1 64->64)
static constexpr long W_R3C1 = 151552;
static constexpr long W_R3C2 = 188416;
static constexpr long W_R3DS = 225280;  // 4096
static constexpr long W_DC2  = 229376;  // 16384 (4 taps x 4096)
static constexpr long W_DC3  = 245760;  // 65536 (16 taps x 4096)
static constexpr long W_FIN  = 311296;  // 110592 (3x3 192->64)

static void launch_repack(hipStream_t s, const float* src, _Float16* dst,
                          int Cout, int Cin, int KH, int KW, int iohw) {
    int total = KH * KW * (Cin >> 5) * 4 * 512;
    repack_weights_kernel<<<(total + 255) / 256, 256, 0, s>>>(src, dst, Cout, Cin, KH, KW, iohw);
}

static void launch_conv(hipStream_t s,
    const _Float16* in, int Hin, int Win, int Cin,
    const _Float16* wpk, int KH, int KW, int stride, int pad,
    int Hout, int Wout,
    const float* gamma, const float* beta, const float* mean, const float* var,
    const float* bias,
    const _Float16* resid, int resC, int relu,
    _Float16* outh, int outC, int outc0, int OH, int OW,
    int osy, int osx, int oy0, int ox0,
    float* outf)
{
    int npix = Hout * Wout;
    dim3 grid((npix + 63) / 64, Bc);
    if (Cin == 64) {
        conv_wmma_kernel<2><<<grid, 128, 0, s>>>(in, Hin, Win, wpk, KH, KW, stride, pad,
            Hout, Wout, gamma, beta, mean, var, bias, resid, resC, relu,
            outh, outC, outc0, OH, OW, osy, osx, oy0, ox0, outf);
    } else {
        conv_wmma_kernel<6><<<grid, 128, 0, s>>>(in, Hin, Win, wpk, KH, KW, stride, pad,
            Hout, Wout, gamma, beta, mean, var, bias, resid, resC, relu,
            outh, outC, outc0, OH, OW, osy, osx, oy0, ox0, outf);
    }
}

extern "C" void kernel_launch(void* const* d_in, const int* in_sizes, int n_in,
                              void* d_out, int out_size, void* d_ws, size_t ws_size,
                              hipStream_t stream) {
    (void)in_sizes; (void)n_in; (void)out_size; (void)ws_size;

    const float* xyz  = (const float*)d_in[IN_XYZ];
    const float* feat = (const float*)d_in[IN_FEAT];
    const int*   bidx = (const int*)d_in[IN_BIDX];

    char* ws = (char*)d_ws;
    float*    canvasF = (float*)(ws + OFF_CANVAS_F32);
    _Float16* canvasH = (_Float16*)(ws + OFF_CANVAS_F16);
    _Float16* tbuf    = (_Float16*)(ws + OFF_TBUF);
    _Float16* dsbuf   = (_Float16*)(ws + OFF_DSBUF);
    _Float16* out2    = (_Float16*)(ws + OFF_OUT2);
    _Float16* out3    = (_Float16*)(ws + OFF_OUT3);
    _Float16* concat  = (_Float16*)(ws + OFF_CONCAT);
    float*    finalF  = (float*)(ws + OFF_FINAL);
    _Float16* wpk     = (_Float16*)(ws + OFF_WPK);

    // BN pointer helpers: leaves are [beta, gamma, mean, var]
    #define BN_ARGS(base) (const float*)d_in[(base)+1], (const float*)d_in[(base)+0], \
                          (const float*)d_in[(base)+2], (const float*)d_in[(base)+3]

    // --- 0) zero canvas ---
    long ncan = (long)Bc * NYc * NXc * Cc;
    zero_f32_kernel<<<2048, 256, 0, stream>>>(canvasF, ncan);

    // --- 1) pointnet + scatter-max ---
    pointnet_scatter_kernel<<<(Nc + 255) / 256, 256, 0, stream>>>(
        xyz, feat, bidx, (const float*)d_in[IN_PN_W], BN_ARGS(IN_PN_BETA), canvasF);

    // --- 2) canvas f32 -> f16 ---
    f32_to_f16_kernel<<<4096, 256, 0, stream>>>(canvasF, canvasH, ncan);

    // --- 3) repack all weights into WMMA B-fragment order ---
    launch_repack(stream, (const float*)d_in[IN_R1_C1], wpk + W_R1C1, 64, 64, 3, 3, 0);
    launch_repack(stream, (const float*)d_in[IN_R1_C2], wpk + W_R1C2, 64, 64, 3, 3, 0);
    launch_repack(stream, (const float*)d_in[IN_R2_C1], wpk + W_R2C1, 64, 64, 3, 3, 0);
    launch_repack(stream, (const float*)d_in[IN_R2_C2], wpk + W_R2C2, 64, 64, 3, 3, 0);
    launch_repack(stream, (const float*)d_in[IN_R2_DS], wpk + W_R2DS, 64, 64, 1, 1, 0);
    launch_repack(stream, (const float*)d_in[IN_R3_C1], wpk + W_R3C1, 64, 64, 3, 3, 0);
    launch_repack(stream, (const float*)d_in[IN_R3_C2], wpk + W_R3C2, 64, 64, 3, 3, 0);
    launch_repack(stream, (const float*)d_in[IN_R3_DS], wpk + W_R3DS, 64, 64, 1, 1, 0);
    launch_repack(stream, (const float*)d_in[IN_DC2_W], wpk + W_DC2, 64, 64, 2, 2, 1); // coij
    launch_repack(stream, (const float*)d_in[IN_DC3_W], wpk + W_DC3, 64, 64, 4, 4, 1); // coij
    launch_repack(stream, (const float*)d_in[IN_CONV_W], wpk + W_FIN, 64, 192, 3, 3, 0);

    // --- 4) res1 (stride 1, identity residual), output -> concat channels 0..63 ---
    launch_conv(stream, canvasH, 248, 216, 64, wpk + W_R1C1, 3, 3, 1, 1, 248, 216,
                BN_ARGS(IN_R1_BN1), nullptr, nullptr, 64, 1,
                tbuf, 64, 0, 248, 216, 1, 1, 0, 0, nullptr);
    launch_conv(stream, tbuf, 248, 216, 64, wpk + W_R1C2, 3, 3, 1, 1, 248, 216,
                BN_ARGS(IN_R1_BN2), nullptr, canvasH, 64, 1,
                concat, 192, 0, 248, 216, 1, 1, 0, 0, nullptr);

    // --- 5) res2 (stride 2, downsample residual) ---
    launch_conv(stream, canvasH, 248, 216, 64, wpk + W_R2DS, 1, 1, 2, 0, 124, 108,
                BN_ARGS(IN_R2_DSBN), nullptr, nullptr, 64, 0,
                dsbuf, 64, 0, 124, 108, 1, 1, 0, 0, nullptr);
    launch_conv(stream, canvasH, 248, 216, 64, wpk + W_R2C1, 3, 3, 2, 1, 124, 108,
                BN_ARGS(IN_R2_BN1), nullptr, nullptr, 64, 1,
                tbuf, 64, 0, 124, 108, 1, 1, 0, 0, nullptr);
    launch_conv(stream, tbuf, 124, 108, 64, wpk + W_R2C2, 3, 3, 1, 1, 124, 108,
                BN_ARGS(IN_R2_BN2), nullptr, dsbuf, 64, 1,
                out2, 64, 0, 124, 108, 1, 1, 0, 0, nullptr);

    // --- 6) res3 (stride 2 from out2) ---
    launch_conv(stream, out2, 124, 108, 64, wpk + W_R3DS, 1, 1, 2, 0, 62, 54,
                BN_ARGS(IN_R3_DSBN), nullptr, nullptr, 64, 0,
                dsbuf, 64, 0, 62, 54, 1, 1, 0, 0, nullptr);
    launch_conv(stream, out2, 124, 108, 64, wpk + W_R3C1, 3, 3, 2, 1, 62, 54,
                BN_ARGS(IN_R3_BN1), nullptr, nullptr, 64, 1,
                tbuf, 64, 0, 62, 54, 1, 1, 0, 0, nullptr);
    launch_conv(stream, tbuf, 62, 54, 64, wpk + W_R3C2, 3, 3, 1, 1, 62, 54,
                BN_ARGS(IN_R3_BN2), nullptr, dsbuf, 64, 1,
                out3, 64, 0, 62, 54, 1, 1, 0, 0, nullptr);

    // --- 7) deconv2: 4 sub-taps (i,j) of a 2x2 kernel -> concat channels 64..127 ---
    for (int i = 0; i < 2; ++i)
        for (int j = 0; j < 2; ++j)
            launch_conv(stream, out2, 124, 108, 64, wpk + W_DC2 + (long)(i * 2 + j) * 4096,
                        1, 1, 1, 0, 124, 108,
                        nullptr, nullptr, nullptr, nullptr, nullptr, nullptr, 64, 0,
                        concat, 192, 64, 248, 216, 2, 2, i, j, nullptr);

    // --- 8) deconv3: 16 sub-taps of a 4x4 kernel -> concat channels 128..191 ---
    for (int i = 0; i < 4; ++i)
        for (int j = 0; j < 4; ++j)
            launch_conv(stream, out3, 62, 54, 64, wpk + W_DC3 + (long)(i * 4 + j) * 4096,
                        1, 1, 1, 0, 62, 54,
                        nullptr, nullptr, nullptr, nullptr, nullptr, nullptr, 64, 0,
                        concat, 192, 128, 248, 216, 4, 4, i, j, nullptr);

    // --- 9) final 3x3 conv, 192 -> 64, + bias, f32 NHWC output ---
    launch_conv(stream, concat, 248, 216, 192, wpk + W_FIN, 3, 3, 1, 1, 248, 216,
                nullptr, nullptr, nullptr, nullptr, (const float*)d_in[IN_CONV_B],
                nullptr, 64, 0,
                nullptr, 64, 0, 248, 216, 1, 1, 0, 0, finalF);

    // --- 10) bilinear grid-sample at point coords -> d_out (N, 64) f32 ---
    long ntot = (long)Nc * Cc;
    grid_sample_kernel<<<(unsigned)((ntot + 255) / 256), 256, 0, stream>>>(
        xyz, bidx, finalF, (float*)d_out);

    #undef BN_ARGS
}
